// MSGSA_55972013801664
// MI455X (gfx1250) — matrix operand order repaired and checked by reading, not compiled, not run on hardware
//
#include <hip/hip_runtime.h>

typedef __attribute__((ext_vector_type(16))) _Float16 v16h;
typedef __attribute__((ext_vector_type(8)))  float    v8f;
typedef __attribute__((ext_vector_type(4)))  float    f32x4;
typedef __attribute__((ext_vector_type(4)))  unsigned int u32x4;
typedef __attribute__((ext_vector_type(8)))  unsigned int u32x8;

#define B_    8
#define N_    4096
#define S_    1024
#define DF_   64
#define INCH_ 67
#define EPS_  1e-5f

// ---------------------------------------------------------------------------
// Tensor Data Mover: DMA a (rows x cols) fp32 tile, row stride = cols, from
// global memory into LDS. D# built per CDNA5 ISA ch.8 (2 SGPR groups, 2D).
// Issued by one wave; completion via TENSORcnt.
// ---------------------------------------------------------------------------
__device__ __forceinline__ void tdm_load_tile_f32(const float* gsrc, float* lds_dst,
                                                  unsigned rows, unsigned cols)
{
    const unsigned long long ga = (unsigned long long)(uintptr_t)gsrc;
    const unsigned la = (unsigned)(uintptr_t)lds_dst; // flat low 32 bits == LDS offset
    u32x4 g0;
    g0[0] = 1u;                                            // count=1, user mode
    g0[1] = la;                                            // lds_addr
    g0[2] = (unsigned)ga;                                  // global_addr[31:0]
    g0[3] = (unsigned)((ga >> 32) & 0x01FFFFFFu) | (2u << 30); // global_addr[56:32], type=2
    u32x8 g1;
    g1[0] = (2u << 16);                                    // data_size=2 (4 bytes)
    g1[1] = (cols & 0xFFFFu) << 16;                        // tensor_dim0[15:0]
    g1[2] = ((cols >> 16) & 0xFFFFu) | ((rows & 0xFFFFu) << 16); // dim0 hi | dim1 lo
    g1[3] = ((rows >> 16) & 0xFFFFu) | ((cols & 0xFFFFu) << 16); // dim1 hi | tile_dim0
    g1[4] = rows & 0xFFFFu;                                // tile_dim1 (tile_dim2=0)
    g1[5] = cols;                                          // tensor_dim0_stride[31:0]
    g1[6] = 0u;
    g1[7] = 0u;
    asm volatile("tensor_load_to_lds %0, %1" :: "s"(g0), "s"(g1) : "memory");
    asm volatile("s_wait_tensorcnt 0x0" ::: "memory");
}

// ---------------------------------------------------------------------------
// Farthest point sampling: one workgroup per batch. Points in LDS, per-thread
// running min-distances in registers, LDS tree argmax (first-index tie-break).
// ---------------------------------------------------------------------------
__global__ __launch_bounds__(1024) void fps_kernel(const float* __restrict__ bpc,
                                                   const int*   __restrict__ far0,
                                                   float* __restrict__ cent,   // (B,S,3)
                                                   float* __restrict__ out)    // fps_pts (B,3,S)
{
    __shared__ float px[N_], py[N_], pz[N_];
    __shared__ float rv[1024];
    __shared__ int   ri[1024];
    const int b = blockIdx.x;
    const int t = threadIdx.x;
    const float* pb = bpc + (size_t)b * 3 * N_;
    for (int i = t; i < N_; i += 1024) {
        px[i] = pb[i];
        py[i] = pb[N_ + i];
        pz[i] = pb[2 * N_ + i];
    }
    __syncthreads();
    float d0 = 1e10f, d1 = 1e10f, d2 = 1e10f, d3 = 1e10f;
    int far = far0[b];
    const int i0 = t, i1 = t + 1024, i2 = t + 2048, i3 = t + 3072;
    for (int s = 0; s < S_; ++s) {
        const float cx = px[far], cy = py[far], cz = pz[far];
        if (t == 0) {
            cent[((size_t)b * S_ + s) * 3 + 0] = cx;
            cent[((size_t)b * S_ + s) * 3 + 1] = cy;
            cent[((size_t)b * S_ + s) * 3 + 2] = cz;
            out[((size_t)b * 3 + 0) * S_ + s] = cx;
            out[((size_t)b * 3 + 1) * S_ + s] = cy;
            out[((size_t)b * 3 + 2) * S_ + s] = cz;
        }
        {
            float dx = px[i0]-cx, dy = py[i0]-cy, dz = pz[i0]-cz;
            d0 = fminf(d0, dx*dx + dy*dy + dz*dz);
            dx = px[i1]-cx; dy = py[i1]-cy; dz = pz[i1]-cz;
            d1 = fminf(d1, dx*dx + dy*dy + dz*dz);
            dx = px[i2]-cx; dy = py[i2]-cy; dz = pz[i2]-cz;
            d2 = fminf(d2, dx*dx + dy*dy + dz*dz);
            dx = px[i3]-cx; dy = py[i3]-cy; dz = pz[i3]-cz;
            d3 = fminf(d3, dx*dx + dy*dy + dz*dz);
        }
        float bv = d0; int bi = i0;
        if (d1 > bv) { bv = d1; bi = i1; }
        if (d2 > bv) { bv = d2; bi = i2; }
        if (d3 > bv) { bv = d3; bi = i3; }
        rv[t] = bv; ri[t] = bi;
        __syncthreads();
        for (int off = 512; off > 0; off >>= 1) {
            if (t < off) {
                const float ov = rv[t + off]; const int oi = ri[t + off];
                if (ov > rv[t] || (ov == rv[t] && oi < ri[t])) { rv[t] = ov; ri[t] = oi; }
            }
            __syncthreads();
        }
        far = ri[0];
        __syncthreads();
    }
}

// ---------------------------------------------------------------------------
// Ball query: one wave32 per query. Ballot/popc compaction keeps the first k
// in-index-order points with ||p-q||^2 <= r^2, padded with the first index.
// ---------------------------------------------------------------------------
__global__ __launch_bounds__(256) void ball_kernel(const float* __restrict__ bpc,
                                                   const float* __restrict__ cent,
                                                   int* __restrict__ gi,
                                                   float r2, int kg)
{
    const int q    = blockIdx.x * (blockDim.x >> 5) + (threadIdx.x >> 5);
    const int lane = threadIdx.x & 31;
    const int b    = q >> 10;
    const float* pb = bpc + (size_t)b * 3 * N_;
    const float qx = cent[(size_t)q * 3 + 0];
    const float qy = cent[(size_t)q * 3 + 1];
    const float qz = cent[(size_t)q * 3 + 2];
    int cnt = 0;
    int first = -1;
    for (int base = 0; base < N_ && cnt < kg; base += 32) {
        const int j = base + lane;
        const float dx = pb[j] - qx, dy = pb[N_ + j] - qy, dz = pb[2 * N_ + j] - qz;
        const bool pred = (dx*dx + dy*dy + dz*dz) <= r2;
        const unsigned mask = (unsigned)__ballot(pred);
        if (pred) {
            const int pos = cnt + __popc(mask & ((1u << lane) - 1u));
            if (pos < kg) gi[(size_t)q * kg + pos] = j;
        }
        if (first < 0 && mask) first = base + __ffs(mask) - 1;
        cnt += __popc(mask);
    }
    for (int pos = cnt + lane; pos < kg; pos += 32)
        gi[(size_t)q * kg + pos] = first;
}

// ---------------------------------------------------------------------------
// GEMM via V_WMMA_F32_16X16X32_F16, fully specialized at compile time.
// One 16x16 output tile per wave, 8 waves/block, weight tile DMA'd to LDS
// by the Tensor Data Mover. GATHER=1: A = [feats[gi] | pts[gi]-centroid].
// GATHER=0: A = relu(scale*y + shift) of previous layer (vectorized b128).
// ---------------------------------------------------------------------------
template<int CIN, int KPAD, int COUT, int KG, int GATHER>
__global__ __launch_bounds__(256) void gemm_wmma(
    const float* __restrict__ Yin,
    const float* __restrict__ F,     // bpc_features (B,64,N)
    const float* __restrict__ P,     // bpc (B,3,N)
    const float* __restrict__ cent,  // (B,S,3)
    const int*   __restrict__ gi,    // (B,S,KG)
    const float* __restrict__ W,     // (COUT, CIN) row-major
    const float* __restrict__ bias,  // (COUT)
    const float* __restrict__ mu,
    const float* __restrict__ rinv,
    const float* __restrict__ gamma,
    const float* __restrict__ beta,
    float* __restrict__ Yout,        // (M, COUT)
    int M)
{
    __shared__ float WsF[16 * CIN];          // weight tile, fp32, via TDM
    __shared__ float scs[KPAD], shs[KPAD];   // fused norm: scale/shift per channel
    const int tid = threadIdx.x;
    const int n0  = blockIdx.y * 16;

    if ((tid >> 5) == 0)
        tdm_load_tile_f32(W + (size_t)n0 * CIN, WsF, 16u, (unsigned)CIN);
    if (GATHER == 0) {
        for (int c = tid; c < CIN; c += 256) {
            const float scv = gamma[c] * rinv[c];
            scs[c] = scv;
            shs[c] = beta[c] - mu[c] * scv;
        }
    }
    __syncthreads();

    const int wave    = tid >> 5;
    const int lane    = tid & 31;
    const int half_id = lane >> 4;
    const int col     = lane & 15;
    const int m0      = (blockIdx.x * 8 + wave) * 16;
    const int arow    = m0 + col;

    // gather-mode row decode (compile-time eliminated for GATHER==0)
    int g = 0;
    float cx = 0.f, cy = 0.f, cz = 0.f;
    const float* Fb = F;
    const float* Pb = P;
    if (GATHER) {
        const int q = arow / KG;             // KG power of two -> shift
        g = gi[arow];
        cx = cent[(size_t)q * 3 + 0];
        cy = cent[(size_t)q * 3 + 1];
        cz = cent[(size_t)q * 3 + 2];
        const int bidx = q >> 10;
        Fb = F + (size_t)bidx * DF_ * N_;
        Pb = P + (size_t)bidx * 3 * N_;
    }
    const float* Yrow = Yin + (size_t)arow * CIN;

    v8f acc = {};
#pragma unroll
    for (int k0 = 0; k0 < KPAD; k0 += 32) {
        // --- A fragment: 16-bit A 16x32 layout ---
        // lanes 0-15 hold K {k0..k0+7, k0+16..k0+23}; lanes 16-31 the other halves.
        const int ka = k0 + half_id * 8;
        const int kb = k0 + 16 + half_id * 8;
        v16h a = {};
        if (GATHER) {
            if (k0 < 64) {
#pragma unroll
                for (int j = 0; j < 8; ++j) {
                    a[j]     = (_Float16)Fb[(size_t)(ka + j) * N_ + g];
                    a[j + 8] = (_Float16)Fb[(size_t)(kb + j) * N_ + g];
                }
            } else {                         // channels 64..66 = relative xyz
                if (half_id == 0) {
                    a[0] = (_Float16)(Pb[g] - cx);
                    a[1] = (_Float16)(Pb[N_ + g] - cy);
                    a[2] = (_Float16)(Pb[2 * N_ + g] - cz);
                }
            }
        } else {
            const f32x4 p0 = *(const f32x4*)(Yrow + ka);
            const f32x4 p1 = *(const f32x4*)(Yrow + ka + 4);
            const f32x4 p2 = *(const f32x4*)(Yrow + kb);
            const f32x4 p3 = *(const f32x4*)(Yrow + kb + 4);
            const f32x4 s0 = *(const f32x4*)&scs[ka];
            const f32x4 s1 = *(const f32x4*)&scs[ka + 4];
            const f32x4 s2 = *(const f32x4*)&scs[kb];
            const f32x4 s3 = *(const f32x4*)&scs[kb + 4];
            const f32x4 h0 = *(const f32x4*)&shs[ka];
            const f32x4 h1 = *(const f32x4*)&shs[ka + 4];
            const f32x4 h2 = *(const f32x4*)&shs[kb];
            const f32x4 h3 = *(const f32x4*)&shs[kb + 4];
#pragma unroll
            for (int j = 0; j < 4; ++j) {
                a[j]      = (_Float16)fmaxf(fmaf(p0[j], s0[j], h0[j]), 0.f);
                a[j + 4]  = (_Float16)fmaxf(fmaf(p1[j], s1[j], h1[j]), 0.f);
                a[j + 8]  = (_Float16)fmaxf(fmaf(p2[j], s2[j], h2[j]), 0.f);
                a[j + 12] = (_Float16)fmaxf(fmaf(p3[j], s3[j], h3[j]), 0.f);
            }
        }

        // --- B fragment: 16-bit B 32x16 layout ---
        // lanes 0-15 hold K {k0..k0+7, k0+16..k0+23} for N=lane; lanes 16-31 the other halves.
        const int kbb = k0 + half_id * 8;
        v16h bv = {};
#pragma unroll
        for (int j = 0; j < 16; ++j) {
            const int c = kbb + (j < 8 ? j : j + 8);
            if (c < CIN) bv[j] = (_Float16)WsF[col * CIN + c];
        }

        acc = __builtin_amdgcn_wmma_f32_16x16x32_f16(
            false, a, false, bv, (short)0, acc, false, false);
    }

    // 32-bit C/D 16x16 layout: VGPR r -> row m0+r (lanes 0-15) / m0+8+r (16-31)
#pragma unroll
    for (int r = 0; r < 8; ++r) {
        const int row = m0 + half_id * 8 + r;
        Yout[(size_t)row * COUT + (n0 + col)] = acc[r] + bias[n0 + col];
    }
    (void)M;
}

// ---------------------------------------------------------------------------
// Per-channel batch statistics over all M rows: mu and rsqrt(var+eps).
// ---------------------------------------------------------------------------
__global__ __launch_bounds__(256) void stats_kernel(const float* __restrict__ Y,
                                                    int M, int C,
                                                    float* __restrict__ mu,
                                                    float* __restrict__ rinv)
{
    __shared__ float ss[256], sq[256];
    const int c = blockIdx.x;
    const int t = threadIdx.x;
    float s = 0.f, s2 = 0.f;
    for (int r = t; r < M; r += 256) {
        const float v = Y[(size_t)r * C + c];
        s += v; s2 += v * v;
    }
    ss[t] = s; sq[t] = s2;
    __syncthreads();
    for (int off = 128; off > 0; off >>= 1) {
        if (t < off) { ss[t] += ss[t + off]; sq[t] += sq[t + off]; }
        __syncthreads();
    }
    if (t == 0) {
        const float m   = ss[0] / (float)M;
        const float var = sq[0] / (float)M - m * m;
        mu[c]   = m;
        rinv[c] = rsqrtf(var + EPS_);
    }
}

// ---------------------------------------------------------------------------
// Final norm + ReLU + max over k, writing (B,320,S) channel block [coff,coff+C)
// ---------------------------------------------------------------------------
__global__ __launch_bounds__(128) void pool_kernel(const float* __restrict__ Y,
                                                   const float* __restrict__ mu,
                                                   const float* __restrict__ rinv,
                                                   const float* __restrict__ gamma,
                                                   const float* __restrict__ beta,
                                                   int kg, int C, int coff,
                                                   float* __restrict__ outF)
{
    const int s = blockIdx.x, b = blockIdx.y;
    const int c = threadIdx.x;
    if (c >= C) return;
    const size_t base = (size_t)(b * S_ + s) * kg;
    const float gm = gamma[c], mm = mu[c], rv = rinv[c], bt = beta[c];
    float m = -1e30f;
    for (int kk = 0; kk < kg; ++kk) {
        const float v = Y[(base + kk) * C + c];
        float a = gm * (v - mm) * rv + bt;
        a = a > 0.f ? a : 0.f;
        if (a > m) m = a;
    }
    outF[((size_t)b * 320 + coff + c) * S_ + s] = m;
}

// ---------------------------------------------------------------------------
// Host orchestration. Workspace (fp32 units):
//   [0, 24576) centroids | [24576,+3*524288) ball idx | [1597440,+128) mu
//   [1597568,+128) rinv | [1597696,+2*67108864) ping/pong activations (~543MB)
// ---------------------------------------------------------------------------
template<int CIN, int KPAD, int COUT, int KG, int GATHER>
static inline void run_gemm(const float* Yin, const float* F, const float* P,
                            const float* cent, const int* gi,
                            const float* W, const float* bias,
                            const float* mu, const float* rinv,
                            const float* gamma, const float* beta,
                            float* Yout, int M, hipStream_t stream)
{
    dim3 grid(M / 128, COUT / 16);
    gemm_wmma<CIN, KPAD, COUT, KG, GATHER><<<grid, 256, 0, stream>>>(
        Yin, F, P, cent, gi, W, bias, mu, rinv, gamma, beta, Yout, M);
}

extern "C" void kernel_launch(void* const* d_in, const int* in_sizes, int n_in,
                              void* d_out, int out_size, void* d_ws, size_t ws_size,
                              hipStream_t stream)
{
    (void)in_sizes; (void)n_in; (void)out_size; (void)ws_size;
    const float* bpc  = (const float*)d_in[0];
    const float* bpcf = (const float*)d_in[1];
    const int*   far0 = (const int*)d_in[2];
    auto WP = [&](int i, int j, int t) { return (const float*)d_in[3 + (i * 3 + j) * 4 + t]; };

    float* out     = (float*)d_out;
    float* ws      = (float*)d_ws;
    float* cent    = ws;
    int*   gi_base = (int*)(ws + 24576);
    float* muB     = ws + 1597440;
    float* rinvB   = ws + 1597568;
    float* bufA    = ws + 1597696;
    float* bufB    = bufA + (size_t)67108864;
    float* featOut = out + (size_t)B_ * 3 * S_;

    fps_kernel<<<B_, 1024, 0, stream>>>(bpc, far0, cent, out);

    // ---- scale 0: r=0.1, k=16, 67->32->32->64 ----
    {
        int* gi = gi_base;
        const int M = B_ * S_ * 16;
        ball_kernel<<<(B_ * S_) / 8, 256, 0, stream>>>(bpc, cent, gi, 0.01f, 16);
        run_gemm<67, 96, 32, 16, 1>(nullptr, bpcf, bpc, cent, gi, WP(0,0,0), WP(0,0,1),
                                    muB, rinvB, nullptr, nullptr, bufA, M, stream);
        stats_kernel<<<32, 256, 0, stream>>>(bufA, M, 32, muB, rinvB);
        run_gemm<32, 32, 32, 16, 0>(bufA, bpcf, bpc, cent, gi, WP(0,1,0), WP(0,1,1),
                                    muB, rinvB, WP(0,0,2), WP(0,0,3), bufB, M, stream);
        stats_kernel<<<32, 256, 0, stream>>>(bufB, M, 32, muB, rinvB);
        run_gemm<32, 32, 64, 16, 0>(bufB, bpcf, bpc, cent, gi, WP(0,2,0), WP(0,2,1),
                                    muB, rinvB, WP(0,1,2), WP(0,1,3), bufA, M, stream);
        stats_kernel<<<64, 256, 0, stream>>>(bufA, M, 64, muB, rinvB);
        pool_kernel<<<dim3(S_, B_), 128, 0, stream>>>(bufA, muB, rinvB, WP(0,2,2), WP(0,2,3),
                                                      16, 64, 0, featOut);
    }
    // ---- scale 1: r=0.2, k=32, 67->64->64->128 ----
    {
        int* gi = gi_base + 524288;
        const int M = B_ * S_ * 32;
        ball_kernel<<<(B_ * S_) / 8, 256, 0, stream>>>(bpc, cent, gi, 0.04f, 32);
        run_gemm<67, 96, 64, 32, 1>(nullptr, bpcf, bpc, cent, gi, WP(1,0,0), WP(1,0,1),
                                    muB, rinvB, nullptr, nullptr, bufA, M, stream);
        stats_kernel<<<64, 256, 0, stream>>>(bufA, M, 64, muB, rinvB);
        run_gemm<64, 64, 64, 32, 0>(bufA, bpcf, bpc, cent, gi, WP(1,1,0), WP(1,1,1),
                                    muB, rinvB, WP(1,0,2), WP(1,0,3), bufB, M, stream);
        stats_kernel<<<64, 256, 0, stream>>>(bufB, M, 64, muB, rinvB);
        run_gemm<64, 64, 128, 32, 0>(bufB, bpcf, bpc, cent, gi, WP(1,2,0), WP(1,2,1),
                                     muB, rinvB, WP(1,1,2), WP(1,1,3), bufA, M, stream);
        stats_kernel<<<128, 256, 0, stream>>>(bufA, M, 128, muB, rinvB);
        pool_kernel<<<dim3(S_, B_), 128, 0, stream>>>(bufA, muB, rinvB, WP(1,2,2), WP(1,2,3),
                                                      32, 128, 64, featOut);
    }
    // ---- scale 2: r=0.4, k=64, 67->64->96->128 ----
    {
        int* gi = gi_base + 2 * 524288;
        const int M = B_ * S_ * 64;
        ball_kernel<<<(B_ * S_) / 8, 256, 0, stream>>>(bpc, cent, gi, 0.16f, 64);
        run_gemm<67, 96, 64, 64, 1>(nullptr, bpcf, bpc, cent, gi, WP(2,0,0), WP(2,0,1),
                                    muB, rinvB, nullptr, nullptr, bufA, M, stream);
        stats_kernel<<<64, 256, 0, stream>>>(bufA, M, 64, muB, rinvB);
        run_gemm<64, 64, 96, 64, 0>(bufA, bpcf, bpc, cent, gi, WP(2,1,0), WP(2,1,1),
                                    muB, rinvB, WP(2,0,2), WP(2,0,3), bufB, M, stream);
        stats_kernel<<<96, 256, 0, stream>>>(bufB, M, 96, muB, rinvB);
        run_gemm<96, 96, 128, 64, 0>(bufB, bpcf, bpc, cent, gi, WP(2,2,0), WP(2,2,1),
                                     muB, rinvB, WP(2,1,2), WP(2,1,3), bufA, M, stream);
        stats_kernel<<<128, 256, 0, stream>>>(bufA, M, 128, muB, rinvB);
        pool_kernel<<<dim3(S_, B_), 128, 0, stream>>>(bufA, muB, rinvB, WP(2,2,2), WP(2,2,3),
                                                      64, 128, 128, featOut);
    }
}
